// _DVNCCodebook_85057532330647
// MI455X (gfx1250) — matrix operand order repaired and checked by reading, not compiled, not run on hardware
//
#include <hip/hip_runtime.h>
#include <hip/hip_bf16.h>
#include <stdint.h>

// Problem sizes (fixed by the reference)
#define D_DIM     512
#define C_CODES   1024
#define KC        32                  // WMMA K per step (bf16 16x16x32)
#define NKSTEPS   (D_DIM / KC)        // 16
#define ROWS_TOTAL 65536              // B*K = 4096*16
#define MTILE     32                  // rows per workgroup (2 WMMA m-tiles)
#define NWGS      (ROWS_TOTAL / MTILE)
#define HN        (ROWS_TOTAL * D_DIM)
#define LN_EPS    1e-5f
#define LOSS_SCALE (1.25f / ((float)ROWS_TOTAL * (float)D_DIM))

typedef __attribute__((ext_vector_type(16))) __bf16 v16bf;
typedef __attribute__((ext_vector_type(8)))  float  v8f;

union Frag16 {
  v16bf v;
  unsigned short u[16];
  uint4 q[2];
};

__device__ __forceinline__ unsigned short f2bf(float f) {
  unsigned int x = __float_as_uint(f);
  x += 0x7FFFu + ((x >> 16) & 1u);        // round-to-nearest-even
  return (unsigned short)(x >> 16);
}

// monotonic float -> uint mapping for atomic-min argmin keys
__device__ __forceinline__ unsigned int orderbits(float f) {
  unsigned int b = __float_as_uint(f);
  return (b & 0x80000000u) ? ~b : (b | 0x80000000u);
}

// ---- WMMA fragment layouts [ISA 7.12.2] --------------------------------
// 16-bit A (16x32, MxK): lanes 0-15: M=lane, VGPR0-3: K=0..7, VGPR4-7: K=16..23
//                        lanes16-31: M=l-16, VGPR0-3: K=8..15, VGPR4-7: K=24..31
// => element e of lane: k = ((e>>3)<<4) + (hi<<3) + (e&7)
// 16-bit B (32x16, KxN): lane<16: N=lane,K=0..7,16..23 ; lane>=16: N=l-16,K=8..15,24..31
// => per lane: two 8-bf16 runs along K (same k order as A).

__device__ __forceinline__ v16bf buildA_bf16row(const unsigned short* row, int lane) {
  const int hi = (lane >> 4) & 1;
  Frag16 f;
#pragma unroll
  for (int e = 0; e < 16; ++e) {
    int kl = ((e >> 3) << 4) + (hi << 3) + (e & 7);
    f.u[e] = row[kl];
  }
  return f.v;
}

// B-fragment: B[k][n] = SRC[n0+n][k0+k], SRC row-major bf16 (K contiguous)
__device__ __forceinline__ v16bf buildB_bf16(const unsigned short* src, int lane) {
  const int n   = lane & 15;
  const int khi = (lane >> 4) << 3;
  const unsigned short* p = src + (size_t)n * D_DIM + khi;
  Frag16 f;
  f.q[0] = ((const uint4*)p)[0];
  f.q[1] = ((const uint4*)p)[2];
  return f.v;
}

// A-fragment from the pre-staged LDS fragment buffer: 32B contiguous per lane
__device__ __forceinline__ v16bf loadA_frag(const unsigned short* p) {
  Frag16 f;
  f.q[0] = ((const uint4*)p)[0];
  f.q[1] = ((const uint4*)p)[1];
  return f.v;
}

__device__ __forceinline__ v8f wmma_bf16(v16bf a, v16bf b, v8f c) {
  return __builtin_amdgcn_wmma_f32_16x16x32_bf16(
      false, a, false, b, (short)0, c, false, false);
}

// Convert an MTILE x D fp32 LDS tile into WMMA A-layout bf16 fragments.
// frag layout: [mt][ks][lane][e] with 32B per (ks,lane) -> single b128 pair loads.
__device__ __forceinline__ void stage_frags(const float* tile,
                                            unsigned short* frag, int tid) {
  for (int it = tid; it < 2 * NKSTEPS * 32; it += 256) {
    int mt = it >> 9;
    int ks = (it >> 5) & (NKSTEPS - 1);
    int lt = it & 31;
    int mhi = lt >> 4, m = mt * 16 + (lt & 15);
    const float* src = tile + m * D_DIM + ks * KC;
    unsigned int* dst = (unsigned int*)(frag + (((mt * NKSTEPS + ks) * 32) + lt) * 16);
#pragma unroll
    for (int e2 = 0; e2 < 8; ++e2) {
      int e = e2 * 2;
      int kl = ((e >> 3) << 4) + (mhi << 3) + (e & 7);
      unsigned int lo   = f2bf(src[kl]);
      unsigned int hi16 = f2bf(src[kl + 1]);
      dst[e2] = lo | (hi16 << 16);
    }
  }
}

// ------------------------------------------------------------------------
// K0: convert W_in / W_out / codebook to bf16 in workspace; codebook norms.
// ------------------------------------------------------------------------
__global__ __launch_bounds__(256)
void vq_prep(const float* __restrict__ Win, const float* __restrict__ Wout,
             const float* __restrict__ cb,
             unsigned short* __restrict__ win_bf, unsigned short* __restrict__ wout_bf,
             unsigned short* __restrict__ cb_bf, float* __restrict__ cnorm) {
  if (blockIdx.x < 1024) {
#pragma unroll
    for (int j = 0; j < 4; ++j) {
      int i = blockIdx.x * 1024 + j * 256 + threadIdx.x;   // [0, 1048576)
      if (i < 262144)            win_bf[i]           = f2bf(Win[i]);
      else if (i < 524288)       wout_bf[i - 262144] = f2bf(Wout[i - 262144]);
      else                       cb_bf[i - 524288]   = f2bf(cb[i - 524288]);
    }
  } else {
    int c = (blockIdx.x - 1024) * 256 + threadIdx.x;       // 4 blocks -> 1024 codes
    const float* row = cb + (size_t)c * D_DIM;
    float s = 0.f;
#pragma unroll 8
    for (int k = 0; k < D_DIM; ++k) { float v = row[k]; s += v * v; }
    cnorm[c] = s;
  }
}

// ------------------------------------------------------------------------
// K1: fused GEMM1 -> dist/argmin -> vq partial -> GEMM3 -> residual+LN.
// One WG (256 thr / 8 wave32) handles 32 rows (2 WMMA m-tiles).
// LDS: 64KB tile (hidden -> z -> out) + 32KB A-fragment buffer (~96.5KB of
// the 320KB WGP LDS -> 3 WGs/WGP).
// ------------------------------------------------------------------------
__global__ __launch_bounds__(256)
void vq_main(const float* __restrict__ hidden, const unsigned char* __restrict__ amask,
             const float* __restrict__ codebook, const float* __restrict__ gamma,
             const float* __restrict__ beta,
             const unsigned short* __restrict__ win_bf,
             const unsigned short* __restrict__ wout_bf,
             const unsigned short* __restrict__ cb_bf,
             const float* __restrict__ cnorm,
             float* __restrict__ partials, float* __restrict__ outp) {
  __shared__ __align__(32) float ldsTile[MTILE][D_DIM];                 // 64KB
  __shared__ __align__(32) unsigned short ldsAfrag[2][NKSTEPS][32][16]; // 32KB
  __shared__ unsigned long long ldsMin[MTILE];
  __shared__ int   ldsIdx[MTILE];
  __shared__ float ldsRed[8];

  const int wg   = blockIdx.x;
  const int m0   = wg * MTILE;
  const int wave = threadIdx.x >> 5;
  const int lane = threadIdx.x & 31;
  const int hi   = lane >> 4;

  if (threadIdx.x < MTILE) ldsMin[threadIdx.x] = ~0ull;

  // warm L2 for the weight streams (global_prefetch_b8)
  if (lane == 0) {
    __builtin_prefetch(win_bf  + (size_t)wave * 64 * D_DIM, 0, 0);
    __builtin_prefetch(cb_bf   + (size_t)wave * 128 * D_DIM, 0, 0);
    __builtin_prefetch(wout_bf + (size_t)wave * 64 * D_DIM, 0, 0);
  }

  // -------- stage hidden tile (64KB, contiguous) into LDS via async copy
  {
    const char* gbase = (const char*)(hidden + (size_t)m0 * D_DIM);
    unsigned    lbase = (unsigned)(uintptr_t)(&ldsTile[0][0]);  // flat[31:0]=LDS off
#pragma unroll
    for (int i = 0; i < 16; ++i) {
      unsigned off = (unsigned)(i * 256 + threadIdx.x) * 16u;   // 16B per lane-op
      unsigned long long ga = (unsigned long long)(uintptr_t)(gbase + off);
      unsigned la = lbase + off;
      asm volatile("global_load_async_to_lds_b128 %0, %1, off"
                   :: "v"(la), "v"(ga) : "memory");
    }
    asm volatile("s_wait_asynccnt 0x0" ::: "memory");
  }
  __syncthreads();

  // -------- stage A-fragments (bf16, WMMA A layout) for GEMM1
  stage_frags(&ldsTile[0][0], &ldsAfrag[0][0][0][0], threadIdx.x);
  __syncthreads();

  // ---------------- GEMM1: z = hidden_tile @ W_in^T  (N split over waves)
  {
    v8f acc[4][2] = {};
#pragma unroll 1
    for (int ks = 0; ks < NKSTEPS; ++ks) {
      v16bf a0 = loadA_frag(&ldsAfrag[0][ks][lane][0]);
      v16bf a1 = loadA_frag(&ldsAfrag[1][ks][lane][0]);
#pragma unroll
      for (int t = 0; t < 4; ++t) {
        int n0 = (wave * 4 + t) * 16;
        v16bf b = buildB_bf16(win_bf + (size_t)n0 * D_DIM + ks * KC, lane);
        acc[t][0] = wmma_bf16(a0, b, acc[t][0]);
        acc[t][1] = wmma_bf16(a1, b, acc[t][1]);
      }
    }
    __syncthreads();   // all frag reads done; hid region now dead -> store z
#pragma unroll
    for (int t = 0; t < 4; ++t) {
      int n0 = (wave * 4 + t) * 16;
#pragma unroll
      for (int mt = 0; mt < 2; ++mt)
#pragma unroll
        for (int r = 0; r < 8; ++r)
          ldsTile[mt * 16 + r + 8 * hi][n0 + (lane & 15)] = acc[t][mt][r];
    }
  }
  __syncthreads();

  // -------- restage A-fragments from z (for the distance GEMM)
  stage_frags(&ldsTile[0][0], &ldsAfrag[0][0][0][0], threadIdx.x);
  __syncthreads();

  // ---------------- dist GEMM: s = z @ codebook^T; argmin via ds_min_u64
#pragma unroll 1
  for (int g = 0; g < 2; ++g) {
    v8f acc[4][2] = {};
#pragma unroll 1
    for (int ks = 0; ks < NKSTEPS; ++ks) {
      v16bf a0 = loadA_frag(&ldsAfrag[0][ks][lane][0]);
      v16bf a1 = loadA_frag(&ldsAfrag[1][ks][lane][0]);
#pragma unroll
      for (int t = 0; t < 4; ++t) {
        int ct = wave * 8 + g * 4 + t;
        v16bf b = buildB_bf16(cb_bf + (size_t)(ct * 16) * D_DIM + ks * KC, lane);
        acc[t][0] = wmma_bf16(a0, b, acc[t][0]);
        acc[t][1] = wmma_bf16(a1, b, acc[t][1]);
      }
    }
#pragma unroll
    for (int t = 0; t < 4; ++t) {
      int ct = wave * 8 + g * 4 + t;
      int c  = ct * 16 + (lane & 15);
      float cn = cnorm[c];
#pragma unroll
      for (int mt = 0; mt < 2; ++mt)
#pragma unroll
        for (int r = 0; r < 8; ++r) {
          float score = cn - 2.0f * acc[t][mt][r];   // ||c||^2 - 2 z.c
          int m = mt * 16 + r + 8 * hi;
          unsigned long long key =
              ((unsigned long long)orderbits(score) << 32) | (unsigned int)c;
          atomicMin(&ldsMin[m], key);                // ds_min_u64, order-independent
        }
    }
  }
  __syncthreads();
  if (threadIdx.x < MTILE)
    ldsIdx[threadIdx.x] = (int)(ldsMin[threadIdx.x] & 0xFFFFFFFFull);
  __syncthreads();

  // ---------------- vq loss partial: sum (z - codebook[idx])^2 over tile
  {
    float s = 0.f;
    for (int i = threadIdx.x; i < MTILE * D_DIM; i += 256) {
      int m = i >> 9, k = i & (D_DIM - 1);
      float d = ldsTile[m][k] - codebook[(size_t)ldsIdx[m] * D_DIM + k];
      s += d * d;
    }
#pragma unroll
    for (int off = 16; off >= 1; off >>= 1) s += __shfl_xor(s, off, 32);
    if (lane == 0) ldsRed[wave] = s;
    __syncthreads();
    if (threadIdx.x == 0) {
      float t = 0.f;
#pragma unroll
      for (int w = 0; w < 8; ++w) t += ldsRed[w];
      partials[wg] = t;
    }
  }
  __syncthreads();   // z dead -> ldsTile reused for `out`

  // ---------------- GEMM3: out = z_q @ W_out^T (z_q gathered bf16 rows)
  {
    v8f acc[4][2] = {};
#pragma unroll 1
    for (int ks = 0; ks < NKSTEPS; ++ks) {
      v16bf a0 = buildA_bf16row(
          cb_bf + (size_t)ldsIdx[lane & 15] * D_DIM + ks * KC, lane);
      v16bf a1 = buildA_bf16row(
          cb_bf + (size_t)ldsIdx[16 + (lane & 15)] * D_DIM + ks * KC, lane);
#pragma unroll
      for (int t = 0; t < 4; ++t) {
        int n0 = (wave * 4 + t) * 16;
        v16bf b = buildB_bf16(wout_bf + (size_t)n0 * D_DIM + ks * KC, lane);
        acc[t][0] = wmma_bf16(a0, b, acc[t][0]);
        acc[t][1] = wmma_bf16(a1, b, acc[t][1]);
      }
    }
#pragma unroll
    for (int t = 0; t < 4; ++t) {
      int n0 = (wave * 4 + t) * 16;
#pragma unroll
      for (int mt = 0; mt < 2; ++mt)
#pragma unroll
        for (int r = 0; r < 8; ++r)
          ldsTile[mt * 16 + r + 8 * hi][n0 + (lane & 15)] = acc[t][mt][r];
    }
  }
  __syncthreads();

  // ---------------- residual + mask + LayerNorm (4 rows per wave)
  // hidden re-read from global: stream is L2-resident (134MB fits 192MB L2)
#pragma unroll 1
  for (int rr = 0; rr < 4; ++rr) {
    int m = wave * 4 + rr;
    size_t rflat = (size_t)m0 + m;
    float af = amask[rflat] ? 1.0f : 0.0f;
    float hv[16], s1 = 0.f, s2 = 0.f;
#pragma unroll
    for (int j = 0; j < 16; ++j) {
      int k = j * 32 + lane;
      float h = hidden[rflat * D_DIM + k] + af * ldsTile[m][k];
      hv[j] = h; s1 += h; s2 += h * h;
    }
#pragma unroll
    for (int off = 16; off >= 1; off >>= 1) {
      s1 += __shfl_xor(s1, off, 32);
      s2 += __shfl_xor(s2, off, 32);
    }
    float mu = s1 * (1.0f / D_DIM);
    float var = s2 * (1.0f / D_DIM) - mu * mu;
    float rs = rsqrtf(var + LN_EPS);
#pragma unroll
    for (int j = 0; j < 16; ++j) {
      int k = j * 32 + lane;
      outp[rflat * D_DIM + k] = (hv[j] - mu) * rs * gamma[k] + beta[k];
    }
  }
}

// ------------------------------------------------------------------------
// K2: deterministic reduction of per-WG loss partials -> d_out[HN]
// ------------------------------------------------------------------------
__global__ __launch_bounds__(256)
void vq_loss_reduce(const float* __restrict__ partials, float* __restrict__ outp) {
  __shared__ float red[256];
  float s = 0.f;
  for (int i = threadIdx.x; i < NWGS; i += 256) s += partials[i];  // fixed order
  red[threadIdx.x] = s;
  __syncthreads();
  for (int off = 128; off >= 1; off >>= 1) {
    if (threadIdx.x < off) red[threadIdx.x] += red[threadIdx.x + off];
    __syncthreads();
  }
  if (threadIdx.x == 0) outp[HN] = red[0] * LOSS_SCALE;
}

// ------------------------------------------------------------------------
extern "C" void kernel_launch(void* const* d_in, const int* in_sizes, int n_in,
                              void* d_out, int out_size, void* d_ws, size_t ws_size,
                              hipStream_t stream) {
  const float*         hidden = (const float*)d_in[0];
  const unsigned char* amask  = (const unsigned char*)d_in[1];  // jax bool = 1 byte
  const float*         Win    = (const float*)d_in[2];
  const float*         Wout   = (const float*)d_in[3];
  const float*         cb     = (const float*)d_in[4];
  const float*         gamma  = (const float*)d_in[5];
  const float*         beta   = (const float*)d_in[6];
  float* outp = (float*)d_out;

  char* ws = (char*)d_ws;
  unsigned short* win_bf  = (unsigned short*)(ws + 0);          // 512 KB
  unsigned short* wout_bf = (unsigned short*)(ws + 524288);     // 512 KB
  unsigned short* cb_bf   = (unsigned short*)(ws + 1048576);    // 1 MB
  float*          cnorm   = (float*)(ws + 2097152);             // 4 KB
  float*          parts   = (float*)(ws + 2101248);             // 8 KB used

  vq_prep<<<1028, 256, 0, stream>>>(Win, Wout, cb, win_bf, wout_bf, cb_bf, cnorm);
  vq_main<<<NWGS, 256, 0, stream>>>(hidden, amask, cb, gamma, beta,
                                    win_bf, wout_bf, cb_bf, cnorm,
                                    parts, outp);
  vq_loss_reduce<<<1, 256, 0, stream>>>(parts, outp);
}
